// TransformerEncoderLayer_4604204941558
// MI455X (gfx1250) — matrix operand order repaired and checked by reading, not compiled, not run on hardware
//
#include <hip/hip_runtime.h>
#include <math.h>

// ---------------- CDNA5 WMMA / TDM types ----------------
typedef __attribute__((ext_vector_type(16))) __bf16 v16bf;
typedef __attribute__((ext_vector_type(8)))  __bf16 v8bf;
typedef __attribute__((ext_vector_type(8)))  float  v8f;
typedef unsigned int u32x4 __attribute__((ext_vector_type(4)));
typedef int          i32x8 __attribute__((ext_vector_type(8)));
typedef int          i32x4 __attribute__((ext_vector_type(4)));

#define TB 256
#define BM 128
#define BN 128
#define BK 32

#if defined(__has_builtin)
#if __has_builtin(__builtin_amdgcn_tensor_load_to_lds) && __has_builtin(__builtin_amdgcn_s_wait_tensorcnt)
#define HAVE_TDM 1
#endif
#endif
#ifndef HAVE_TDM
#define HAVE_TDM 0
#endif

enum { EPI_BIAS_BF16 = 0, EPI_SCALE_F32 = 1, EPI_BIAS_F32 = 2,
       EPI_BIAS_RELU_BF16 = 3, EPI_BF16 = 4 };

union V16U { v16bf v; v8bf h[2]; };
union V8Z  { v8bf v; unsigned u[4]; };

#if HAVE_TDM
// ---------------------------------------------------------------------------
// Issue one TDM 2-D tile load: `rows` x BK bf16 elements, row stride
// `stride_elems`, packed contiguously into LDS at byte offset `lds_off`.
// D# per CDNA5 ISA ch.8: group0 {count, lds_addr, global_addr, type=2},
// group1 {data_size=1(2B), tensor_dim=tile_dim=BKxrows, dim0 stride}.
// This toolchain exposes the 6-arg builtin (extra zero-filled i32x8 group).
// ---------------------------------------------------------------------------
__device__ __forceinline__ void tdm_load_tile(unsigned lds_off,
                                              const __bf16* gptr,
                                              unsigned rows,
                                              unsigned long stride_elems)
{
    unsigned long ga = (unsigned long)gptr;
    u32x4 g0;
    g0[0] = 1u;                                            // count=1 (valid D#)
    g0[1] = lds_off;                                       // LDS byte address
    g0[2] = (unsigned)(ga & 0xFFFFFFFFu);                  // global addr lo
    g0[3] = (unsigned)((ga >> 32) & 0x01FFFFFFu) | (2u << 30); // hi + type=2

    const unsigned td0 = BK;                               // tensor_dim0 (elems)
    const unsigned td1 = rows;                             // tensor_dim1
    i32x8 g1;
    g1[0] = (int)(1u << 16);                               // data_size=1 -> 2B
    g1[1] = (int)((td0 & 0xFFFFu) << 16);                  // tensor_dim0[15:0]
    g1[2] = (int)((td0 >> 16) | ((td1 & 0xFFFFu) << 16));  // dim0 hi | dim1 lo
    g1[3] = (int)((td1 >> 16) | ((unsigned)BK << 16));     // dim1 hi | tile_dim0
    g1[4] = (int)(rows & 0xFFFFu);                         // tile_dim1, tile_dim2=0
    g1[5] = (int)(stride_elems & 0xFFFFFFFFu);             // dim0 stride lo
    g1[6] = (int)((stride_elems >> 32) & 0xFFFFu);         // dim0 stride hi
    g1[7] = 0;                                             // dim1 stride unused (2D)

    i32x4 z4; z4[0] = z4[1] = z4[2] = z4[3] = 0;           // groups 2/3 unused
    i32x8 z8; z8[0] = z8[1] = z8[2] = z8[3] = 0;
    z8[4] = z8[5] = z8[6] = z8[7] = 0;
    __builtin_amdgcn_tensor_load_to_lds(g0, g1, z4, z4, z8, 0);
}
#endif

// ---------------------------------------------------------------------------
// Per-wave WMMA work on one staged K-slab: 2x4 tiles of 16x16, K=32.
// Fragment loads follow the documented CDNA5 wave32 VGPR layouts.
// ---------------------------------------------------------------------------
__device__ __forceinline__ void wmma_tiles(const __bf16* __restrict__ Asrc,
                                           const __bf16* __restrict__ Bsrc,
                                           int wm, int wn, int half, int lrow,
                                           v8f acc[2][4])
{
    v16bf afrag[2];
    #pragma unroll
    for (int mt = 0; mt < 2; ++mt) {
        int r = wm * 32 + mt * 16 + lrow;
        const __bf16* base = &Asrc[r * BK];
        V16U t;
        t.h[0] = *(const v8bf*)(base + 8 * half);       // K = 8*half .. +7
        t.h[1] = *(const v8bf*)(base + 8 * half + 16);  // K = 8*half+16 .. +23
        afrag[mt] = t.v;
    }
    v16bf bfrag[4];
    #pragma unroll
    for (int nt = 0; nt < 4; ++nt) {
        int n = wn * 64 + nt * 16 + lrow;
        const __bf16* base = &Bsrc[n * BK + 8 * half]; // K = 8*half .. +7, 8*half+16 .. +23
        V16U t;
        t.h[0] = *(const v8bf*)(base);
        t.h[1] = *(const v8bf*)(base + 16);
        bfrag[nt] = t.v;
    }
    #pragma unroll
    for (int mt = 0; mt < 2; ++mt)
        #pragma unroll
        for (int nt = 0; nt < 4; ++nt)
            acc[mt][nt] = __builtin_amdgcn_wmma_f32_16x16x32_bf16(
                false, afrag[mt], false, bfrag[nt],
                (short)0, acc[mt][nt], false, false);
}

// ---------------------------------------------------------------------------
// Generic mixed-precision GEMM: C[M,N] = A[M,K] (bf16 row-major, lda)
//                                        x Bt[N,K] (bf16, strides bns/bks)
// f32 accumulation via V_WMMA_F32_16X16X32_BF16, fused epilogue.
// Fast path: TDM double-buffered pipeline (full tiles, contiguous K).
// ---------------------------------------------------------------------------
__global__ __launch_bounds__(256)
void gemm_bf16_wmma(const __bf16* __restrict__ A, int lda,
                    const __bf16* __restrict__ Bt, long bns, long bks,
                    int M, int N, int K,
                    const float* __restrict__ bias, float scale,
                    float* __restrict__ outf, __bf16* __restrict__ outb,
                    int ldo, int epi)
{
    __shared__ __bf16 As[2][BM * BK];   // [m][k], double buffered
    __shared__ __bf16 Bs[2][BN * BK];   // [n][k], double buffered

    const int tid  = threadIdx.x;
    const int lane = tid & 31;
    const int wave = tid >> 5;          // 0..7
    const int wm   = wave & 3;          // 32-row band
    const int wn   = wave >> 2;         // 64-col band
    const int half = lane >> 4;
    const int lrow = lane & 15;

    const int blockN = blockIdx.x * BN;
    const int blockM = blockIdx.y * BM;

    v8f acc[2][4];
    for (int a = 0; a < 2; ++a)
        for (int b = 0; b < 4; ++b)
            for (int i = 0; i < 8; ++i) acc[a][b][i] = 0.0f;

    const int ksteps = K / BK;

#if HAVE_TDM
    const bool full = (bks == 1) && (blockM + BM <= M) && (blockN + BN <= N);
    if (full) {
        // Each wave DMAs its own 16xBK slice of the A and B tiles; waits on
        // its own TENSORcnt before the barrier publishes the full tile.
        const __bf16* Ag = A  + (long)blockM * lda;
        const __bf16* Bg = Bt + (long)blockN * bns;
        const unsigned aoff = wave * 16 * BK * 2;   // bytes within tile
        // prologue: stage k-slab 0 into buffer 0
        tdm_load_tile((unsigned)(unsigned long)(const void*)&As[0][0] + aoff,
                      Ag + (long)(wave * 16) * lda, 16, (unsigned long)lda);
        tdm_load_tile((unsigned)(unsigned long)(const void*)&Bs[0][0] + aoff,
                      Bg + (long)(wave * 16) * bns, 16, (unsigned long)bns);
        __builtin_amdgcn_s_wait_tensorcnt(0);
        __syncthreads();

        for (int kt = 0; kt < ksteps; ++kt) {
            const int cur = kt & 1, nxt = cur ^ 1;
            const bool more = (kt + 1 < ksteps);
            if (more) {                 // async-prefetch next slab (overlaps WMMA)
                const int k1 = (kt + 1) * BK;
                tdm_load_tile((unsigned)(unsigned long)(const void*)&As[nxt][0] + aoff,
                              Ag + (long)(wave * 16) * lda + k1, 16, (unsigned long)lda);
                tdm_load_tile((unsigned)(unsigned long)(const void*)&Bs[nxt][0] + aoff,
                              Bg + (long)(wave * 16) * bns + k1, 16, (unsigned long)bns);
            }
            wmma_tiles(&As[cur][0], &Bs[cur][0], wm, wn, half, lrow, acc);
            if (more) __builtin_amdgcn_s_wait_tensorcnt(0);
            __syncthreads();
        }
    } else
#endif
    {
        // generic path: cooperative staging with bounds guards
        for (int kt = 0; kt < ksteps; ++kt) {
            const int k0 = kt * BK;
            {   // A tile, 16B vector loads
                int r0 = tid >> 2;
                int c  = (tid & 3) * 8;
                for (int p = 0; p < 2; ++p) {
                    int r  = p * 64 + r0;
                    int gm = blockM + r;
                    V8Z val; val.u[0] = val.u[1] = val.u[2] = val.u[3] = 0u;
                    if (gm < M) val.v = *(const v8bf*)(A + (long)gm * lda + (k0 + c));
                    *(v8bf*)&As[0][r * BK + c] = val.v;
                }
            }
            if (bks == 1) {
                int r0 = tid >> 2;
                int c  = (tid & 3) * 8;
                for (int p = 0; p < 2; ++p) {
                    int r  = p * 64 + r0;
                    int gn = blockN + r;
                    V8Z val; val.u[0] = val.u[1] = val.u[2] = val.u[3] = 0u;
                    if (gn < N) val.v = *(const v8bf*)(Bt + (long)gn * bns + (k0 + c));
                    *(v8bf*)&Bs[0][r * BK + c] = val.v;
                }
            } else {                     // strided K (P·V path)
                for (int i = 0; i < 16; ++i) {
                    int idx = i * TB + tid;
                    int c   = idx >> 7;
                    int n   = idx & 127;
                    int gn  = blockN + n;
                    __bf16 v = (__bf16)0.0f;
                    if (gn < N) v = Bt[(long)gn * bns + (long)(k0 + c) * bks];
                    Bs[0][n * BK + c] = v;
                }
            }
            __syncthreads();
            if (kt + 1 < ksteps) {       // L2 prefetch of next A slab
                int r  = tid >> 2;
                int gm = blockM + r;
                if (gm < M) __builtin_prefetch(A + (long)gm * lda + (k0 + BK), 0, 0);
            }
            wmma_tiles(&As[0][0], &Bs[0][0], wm, wn, half, lrow, acc);
            __syncthreads();
        }
    }

    // ---- epilogue: lane holds col N=lrow, VGPR i -> row M = i + 8*half ----
    for (int mt = 0; mt < 2; ++mt) {
        for (int nt = 0; nt < 4; ++nt) {
            v8f c = acc[mt][nt];
            int nn = blockN + wn * 64 + nt * 16 + lrow;
            if (nn >= N) continue;
            for (int i = 0; i < 8; ++i) {
                int mm = blockM + wm * 32 + mt * 16 + 8 * half + i;
                if (mm >= M) continue;
                float v = c[i];
                long  o = (long)mm * ldo + nn;
                switch (epi) {
                  case EPI_BIAS_BF16:      outb[o] = (__bf16)(v + bias[nn]); break;
                  case EPI_SCALE_F32:      outf[o] = v * scale;              break;
                  case EPI_BIAS_F32:       outf[o] = v + bias[nn];           break;
                  case EPI_BIAS_RELU_BF16: { float t = v + bias[nn];
                                             outb[o] = (__bf16)(t > 0.f ? t : 0.f); } break;
                  default:                 outb[o] = (__bf16)v;              break;
                }
            }
        }
    }
}

// ---------------------------------------------------------------------------
// Row softmax: f32 scores -> bf16 probabilities (one block per row).
// ---------------------------------------------------------------------------
__global__ __launch_bounds__(256)
void softmax_rows(const float* __restrict__ s, __bf16* __restrict__ p, int n)
{
    const long base = (long)blockIdx.x * n;
    __shared__ float red[256];
    int tid = threadIdx.x;

    float mx = -1e30f;
    for (int j = tid; j < n; j += 256) mx = fmaxf(mx, s[base + j]);
    red[tid] = mx; __syncthreads();
    for (int st = 128; st > 0; st >>= 1) {
        if (tid < st) red[tid] = fmaxf(red[tid], red[tid + st]);
        __syncthreads();
    }
    mx = red[0]; __syncthreads();

    float sum = 0.f;
    for (int j = tid; j < n; j += 256) sum += __expf(s[base + j] - mx);
    red[tid] = sum; __syncthreads();
    for (int st = 128; st > 0; st >>= 1) {
        if (tid < st) red[tid] += red[tid + st];
        __syncthreads();
    }
    float inv = 1.0f / red[0];

    for (int j = tid; j < n; j += 256)
        p[base + j] = (__bf16)(__expf(s[base + j] - mx) * inv);
}

// ---------------------------------------------------------------------------
// out = LayerNorm(x + r) * g + b ; optional bf16 mirror for the next GEMM.
// ---------------------------------------------------------------------------
__global__ __launch_bounds__(256)
void add_layernorm(const float* __restrict__ x, const float* __restrict__ r,
                   const float* __restrict__ g, const float* __restrict__ b,
                   float* __restrict__ outf, __bf16* __restrict__ outb, int D)
{
    const long base = (long)blockIdx.x * D;
    __shared__ float s1[256], s2[256];
    int tid = threadIdx.x;

    float sum = 0.f, sq = 0.f;
    for (int j = tid; j < D; j += 256) {
        float v = x[base + j] + r[base + j];
        sum += v; sq += v * v;
    }
    s1[tid] = sum; s2[tid] = sq; __syncthreads();
    for (int st = 128; st > 0; st >>= 1) {
        if (tid < st) { s1[tid] += s1[tid + st]; s2[tid] += s2[tid + st]; }
        __syncthreads();
    }
    float mu  = s1[0] / D;
    float var = s2[0] / D - mu * mu;
    float inv = rsqrtf(var + 1e-5f);

    for (int j = tid; j < D; j += 256) {
        float v = x[base + j] + r[base + j];
        float y = (v - mu) * inv * g[j] + b[j];
        outf[base + j] = y;
        if (outb) outb[base + j] = (__bf16)y;
    }
}

// ---------------------------------------------------------------------------
// Packing: f32 -> bf16 (row-major), and f32 [K,N] -> bf16 transposed [N,K].
// ---------------------------------------------------------------------------
__global__ __launch_bounds__(256)
void pack_bf16(const float* __restrict__ in, __bf16* __restrict__ out, long n)
{
    long i = (long)blockIdx.x * 256 + threadIdx.x;
    if (i < n) out[i] = (__bf16)in[i];
}

__global__ __launch_bounds__(256)
void pack_wt_bf16(const float* __restrict__ W, __bf16* __restrict__ Wt,
                  int K, int N)
{
    long i = (long)blockIdx.x * 256 + threadIdx.x;
    long total = (long)K * N;
    if (i < total) {
        long k = i / N, n = i % N;
        Wt[n * (long)K + k] = (__bf16)W[i];
    }
}

// ---------------------------------------------------------------------------
// Host-side orchestration.
// ---------------------------------------------------------------------------
extern "C" void kernel_launch(void* const* d_in, const int* in_sizes, int n_in,
                              void* d_out, int out_size, void* d_ws, size_t ws_size,
                              hipStream_t stream)
{
    (void)in_sizes; (void)n_in; (void)out_size; (void)ws_size;

    const int cB = 2, cS = 2048, cD = 1024, cH = 16, cDK = 64, cFF = 4096;
    const int M = cB * cS;                      // 4096 token rows

    const float* x  = (const float*)d_in[0];
    const float* Wq = (const float*)d_in[1];  const float* bq = (const float*)d_in[2];
    const float* Wk = (const float*)d_in[3];  const float* bk = (const float*)d_in[4];
    const float* Wv = (const float*)d_in[5];  const float* bv = (const float*)d_in[6];
    const float* Wo = (const float*)d_in[7];  const float* bo = (const float*)d_in[8];
    const float* W1 = (const float*)d_in[9];  const float* b1 = (const float*)d_in[10];
    const float* W2 = (const float*)d_in[11]; const float* b2 = (const float*)d_in[12];
    const float* g1 = (const float*)d_in[13]; const float* be1 = (const float*)d_in[14];
    const float* g2 = (const float*)d_in[15]; const float* be2 = (const float*)d_in[16];

    char* ws = (char*)d_ws;
    auto alloc = [&](size_t bytes) -> char* {
        char* p = ws; ws += (bytes + 255) & ~(size_t)255; return p;
    };
    __bf16* xb    = (__bf16*)alloc((size_t)M * cD * 2);
    __bf16* wqt   = (__bf16*)alloc((size_t)cD * cD * 2);
    __bf16* wkt   = (__bf16*)alloc((size_t)cD * cD * 2);
    __bf16* wvt   = (__bf16*)alloc((size_t)cD * cD * 2);
    __bf16* wot   = (__bf16*)alloc((size_t)cD * cD * 2);
    __bf16* w1t   = (__bf16*)alloc((size_t)cFF * cD * 2);   // [N=FF][K=D]
    __bf16* w2t   = (__bf16*)alloc((size_t)cD * cFF * 2);   // [N=D][K=FF]
    __bf16* qb    = (__bf16*)alloc((size_t)M * cD * 2);
    __bf16* kbuf  = (__bf16*)alloc((size_t)M * cD * 2);
    __bf16* vbuf  = (__bf16*)alloc((size_t)M * cD * 2);
    float*  scores= (float*) alloc((size_t)cS * cS * 4);    // reused per (b,h)
    __bf16* probs = (__bf16*)alloc((size_t)cS * cS * 2);    // reused per (b,h)
    __bf16* attnb = (__bf16*)alloc((size_t)M * cD * 2);
    float*  atout = (float*) alloc((size_t)M * cD * 4);
    float*  hres  = (float*) alloc((size_t)M * cD * 4);
    __bf16* hb    = (__bf16*)alloc((size_t)M * cD * 2);
    __bf16* r1    = (__bf16*)alloc((size_t)M * cFF * 2);
    float*  ffn2  = (float*) alloc((size_t)M * cD * 4);

    auto gemm = [&](const __bf16* A, int lda, const __bf16* Bt, long bns, long bks,
                    int gM, int gN, int gK, const float* bias, float scale,
                    float* outf, __bf16* outb, int ldo, int epi) {
        dim3 grid((gN + BN - 1) / BN, (gM + BM - 1) / BM);
        gemm_bf16_wmma<<<grid, dim3(TB), 0, stream>>>(A, lda, Bt, bns, bks,
            gM, gN, gK, bias, scale, outf, outb, ldo, epi);
    };

    // 1) pack activations & weights to bf16 (weights N-major for the B path)
    {
        long n = (long)M * cD;
        pack_bf16<<<dim3((unsigned)((n + 255) / 256)), dim3(256), 0, stream>>>(x, xb, n);
        long wn = (long)cD * cD;
        dim3 wg((unsigned)((wn + 255) / 256));
        pack_wt_bf16<<<wg, dim3(256), 0, stream>>>(Wq, wqt, cD, cD);
        pack_wt_bf16<<<wg, dim3(256), 0, stream>>>(Wk, wkt, cD, cD);
        pack_wt_bf16<<<wg, dim3(256), 0, stream>>>(Wv, wvt, cD, cD);
        pack_wt_bf16<<<wg, dim3(256), 0, stream>>>(Wo, wot, cD, cD);
        long f1 = (long)cD * cFF;
        pack_wt_bf16<<<dim3((unsigned)((f1 + 255) / 256)), dim3(256), 0, stream>>>(W1, w1t, cD, cFF);
        pack_wt_bf16<<<dim3((unsigned)((f1 + 255) / 256)), dim3(256), 0, stream>>>(W2, w2t, cFF, cD);
    }

    // 2) Q/K/V projections (bias fused, bf16 out in [b,s,h,dk] layout)
    gemm(xb, cD, wqt, cD, 1, M, cD, cD, bq, 0.f, nullptr, qb,   cD, EPI_BIAS_BF16);
    gemm(xb, cD, wkt, cD, 1, M, cD, cD, bk, 0.f, nullptr, kbuf, cD, EPI_BIAS_BF16);
    gemm(xb, cD, wvt, cD, 1, M, cD, cD, bv, 0.f, nullptr, vbuf, cD, EPI_BIAS_BF16);

    // 3) attention, staged per (b,h): S = QK^T/8 -> softmax -> P·V
    for (int b = 0; b < cB; ++b) {
        for (int h = 0; h < cH; ++h) {
            long off = (long)b * cS * cD + (long)h * cDK;
            gemm(qb + off, cD, kbuf + off, cD, 1, cS, cS, cDK,
                 nullptr, 0.125f, scores, nullptr, cS, EPI_SCALE_F32);
            softmax_rows<<<dim3(cS), dim3(256), 0, stream>>>(scores, probs, cS);
            gemm(probs, cS, vbuf + off, 1, cD, cS, cDK, cS,
                 nullptr, 0.f, nullptr, attnb + off, cD, EPI_BF16);
        }
    }

    // 4) output projection + residual + LN1
    gemm(attnb, cD, wot, cD, 1, M, cD, cD, bo, 0.f, atout, nullptr, cD, EPI_BIAS_F32);
    add_layernorm<<<dim3(M), dim3(256), 0, stream>>>(x, atout, g1, be1, hres, hb, cD);

    // 5) FFN: relu(h@W1+b1)@W2+b2, then residual + LN2 -> d_out
    gemm(hb, cD, w1t, cD, 1, M, cFF, cD, b1, 0.f, nullptr, r1, cFF, EPI_BIAS_RELU_BF16);
    gemm(r1, cFF, w2t, cFF, 1, M, cD, cFF, b2, 0.f, ffn2, nullptr, cD, EPI_BIAS_F32);
    add_layernorm<<<dim3(M), dim3(256), 0, stream>>>(hres, ffn2, g2, be2,
                                                     (float*)d_out, nullptr, cD);
}
